// MultiHeadAttn_79156247265928
// MI455X (gfx1250) — compile-verified, matrix operand order checked by reading, not run on hardware
//
#include <hip/hip_runtime.h>

#define NH 16
#define DH 64
#define DM 1024
#define SEQ 2048
#define BATCH 2
#define ROWS (SEQ * BATCH) /* 4096 */

typedef __attribute__((ext_vector_type(8)))  __bf16 v8bf;
typedef __attribute__((ext_vector_type(16))) __bf16 v16bf;
typedef __attribute__((ext_vector_type(8)))  float  v8f;

union BF16Frag { v16bf v; v8bf h[2]; };

static __device__ __forceinline__ v8f wmma_bf16(v16bf a, v16bf b, v8f c) {
  // (neg_a, A, neg_b, B, c_mod, C, reuse_a, reuse_b)
  return __builtin_amdgcn_wmma_f32_16x16x32_bf16(false, a, false, b, (short)0, c, false, false);
}

// ---------------- conversion kernels ----------------

__global__ void cvt_f32_bf16(const float* __restrict__ in, __bf16* __restrict__ out, int n) {
  int i = blockIdx.x * blockDim.x + threadIdx.x;
  if (i < n) out[i] = (__bf16)in[i];
}

// W[K,N] f32 -> Wt[N,K] bf16 (so B-fragments load contiguous K-runs per lane)
__global__ void cvt_transpose(const float* __restrict__ W, __bf16* __restrict__ Wt, int K, int N) {
  int i = blockIdx.x * blockDim.x + threadIdx.x; // i = n*K + k
  if (i >= K * N) return;
  int n = i / K, k = i - n * K;
  Wt[i] = (__bf16)W[(size_t)k * N + n];
}

// kvbf[L*B, 2048] (cols 1024.. are V) -> Vt[b, n, d, L]
__global__ void repack_vt(const __bf16* __restrict__ kvbf, __bf16* __restrict__ Vt) {
  int i = blockIdx.x * blockDim.x + threadIdx.x; // [0, ROWS*1024)
  if (i >= ROWS * DM) return;
  int row = i >> 10;  // l*B + b
  int nd  = i & 1023; // n*64 + d
  int l = row >> 1;   // BATCH == 2
  int b = row & 1;
  Vt[((size_t)b * 1024 + nd) * SEQ + l] = kvbf[(size_t)row * 2048 + 1024 + nd];
}

// ---------------- GEMM: C[M,N] = A[M,K] @ Bt[N,K]^T, bf16 in, f32 acc ----------------
// block = 128 threads (4 waves); wave w: rows by*64+w*16, cols bx*64 (4 tiles of 16)

template <bool OUT_BF16>
__global__ void gemm_bf16(const __bf16* __restrict__ A, const __bf16* __restrict__ Bt,
                          void* __restrict__ Cp, int M, int N, int K) {
  const int wave = threadIdx.x >> 5;
  const int lane = threadIdx.x & 31;
  const int r = lane & 15, half = lane >> 4;
  const int m0 = blockIdx.y * 64 + wave * 16;
  const int n0 = blockIdx.x * 64;

  v8f acc[4] = {};
  const __bf16* arow = A + (size_t)(m0 + r) * K;

  for (int k0 = 0; k0 < K; k0 += 32) {
    BF16Frag af;
    af.h[0] = *(const v8bf*)(arow + k0 + half * 8);      // K = k0 + half*8 .. +7
    af.h[1] = *(const v8bf*)(arow + k0 + 16 + half * 8); // K = k0+16+half*8 .. +7
#pragma unroll
    for (int j = 0; j < 4; ++j) {
      const __bf16* brow = Bt + (size_t)(n0 + j * 16 + r) * K + k0;
      BF16Frag bf;
      bf.h[0] = *(const v8bf*)(brow + half * 8);
      bf.h[1] = *(const v8bf*)(brow + 16 + half * 8);
      acc[j] = wmma_bf16(af.v, bf.v, acc[j]);
    }
  }

#pragma unroll
  for (int j = 0; j < 4; ++j) {
    int n = n0 + j * 16 + r;
#pragma unroll
    for (int i = 0; i < 8; ++i) {
      int m = m0 + half * 8 + i; // C layout: vgpr i holds (M = half*8+i, N = lane%16)
      if constexpr (OUT_BF16)
        ((__bf16*)Cp)[(size_t)m * N + n] = (__bf16)acc[j][i];
      else
        ((float*)Cp)[(size_t)m * N + n] = acc[j][i];
    }
  }
}

// ---------------- attention: one block per (16-query tile, b, n) ----------------
// 8 waves. LDS: S[16][2048] f32 (128KB) + P[16][2048] bf16 (64KB) = 192KB (<320KB/WGP).

__global__ void attn_kernel(const __bf16* __restrict__ qbf, const __bf16* __restrict__ kvbf,
                            const __bf16* __restrict__ Vt, __bf16* __restrict__ vecbf) {
  extern __shared__ char smem[];
  float*  s_s = (float*)smem;                       // [16][SEQ] scores
  __bf16* s_p = (__bf16*)(smem + 16 * SEQ * 4);     // [16][SEQ] probabilities

  const int q0 = blockIdx.x * 16;
  const int bn = blockIdx.y;
  const int b = bn >> 4, n = bn & 15;
  const int tid = threadIdx.x;
  const int wave = tid >> 5, lane = tid & 31;
  const int r = lane & 15, half = lane >> 4;

  // Q A-fragments for head-dim chunks {0..31}, {32..63}: lane holds query row q0+r
  const __bf16* qrow = qbf + ((size_t)(q0 + r) * BATCH + b) * DM + n * DH;
  BF16Frag aq[2];
#pragma unroll
  for (int d = 0; d < 2; ++d) {
    aq[d].h[0] = *(const v8bf*)(qrow + d * 32 + half * 8);
    aq[d].h[1] = *(const v8bf*)(qrow + d * 32 + 16 + half * 8);
  }

  // ---- phase 1: S = (Q K^T) * 1/sqrt(64); wave w owns keys [w*256, w*256+256) ----
  for (int t = 0; t < 16; ++t) {
    const int j0 = wave * 256 + t * 16;
    v8f sacc = {};
#pragma unroll
    for (int d = 0; d < 2; ++d) {
      // B-frag: lane column = key j0+r, K dim = head dims (contiguous in memory)
      const __bf16* krow = kvbf + ((size_t)(j0 + r) * BATCH + b) * 2048 + n * DH + d * 32;
      BF16Frag bk;
      bk.h[0] = *(const v8bf*)(krow + half * 8);
      bk.h[1] = *(const v8bf*)(krow + 16 + half * 8);
      sacc = wmma_bf16(aq[d].v, bk.v, sacc);
    }
#pragma unroll
    for (int i = 0; i < 8; ++i)
      s_s[(half * 8 + i) * SEQ + j0 + r] = sacc[i] * 0.125f;
  }
  __syncthreads();

  // ---- phase 2: row softmax (16 lanes per row; butterfly within wave32 halves) ----
  {
    const int rr = tid >> 4, c0 = tid & 15;
    float mx = -3.0e38f;
    for (int c = c0; c < SEQ; c += 16) mx = fmaxf(mx, s_s[rr * SEQ + c]);
#pragma unroll
    for (int off = 8; off >= 1; off >>= 1) mx = fmaxf(mx, __shfl_xor(mx, off, 32));
    float sum = 0.f;
    for (int c = c0; c < SEQ; c += 16) sum += __expf(s_s[rr * SEQ + c] - mx);
#pragma unroll
    for (int off = 8; off >= 1; off >>= 1) sum += __shfl_xor(sum, off, 32);
    const float inv = 1.0f / sum;
    for (int c = c0; c < SEQ; c += 16)
      s_p[rr * SEQ + c] = (__bf16)(__expf(s_s[rr * SEQ + c] - mx) * inv);
  }
  __syncthreads();

  // ---- phase 3: O = P @ V; waves 0..3 own head-dim column tiles of 16 ----
  if (wave < 4) {
    const int d0 = wave * 16;
    v8f o = {};
    const __bf16* vtb = Vt + ((size_t)b * 1024 + n * DH + d0 + r) * SEQ;
    for (int j0 = 0; j0 < SEQ; j0 += 32) {
      BF16Frag ap, bv;
      // A-frag: lane row = query r, K dim = keys (ds reads do the C->A relayout)
      ap.h[0] = *(const v8bf*)(s_p + r * SEQ + j0 + half * 8);
      ap.h[1] = *(const v8bf*)(s_p + r * SEQ + j0 + 16 + half * 8);
      // B-frag: lane column = head dim d0+r, K dim = keys (contiguous via Vt)
      bv.h[0] = *(const v8bf*)(vtb + j0 + half * 8);
      bv.h[1] = *(const v8bf*)(vtb + j0 + 16 + half * 8);
      o = wmma_bf16(ap.v, bv.v, o);
    }
#pragma unroll
    for (int i = 0; i < 8; ++i) {
      int row = q0 + half * 8 + i;
      vecbf[((size_t)row * BATCH + b) * DM + n * DH + d0 + r] = (__bf16)o[i];
    }
  }
}

// ---------------- launch ----------------

extern "C" void kernel_launch(void* const* d_in, const int* in_sizes, int n_in,
                              void* d_out, int out_size, void* d_ws, size_t ws_size,
                              hipStream_t stream) {
  (void)in_sizes; (void)n_in; (void)out_size; (void)ws_size;
  const float* h   = (const float*)d_in[0];
  const float* Wq  = (const float*)d_in[1];
  const float* Wkv = (const float*)d_in[2];
  const float* Wo  = (const float*)d_in[3];

  char* ws = (char*)d_ws;
  size_t off = 0;
  auto alloc = [&](size_t bytes) -> char* {
    char* p = ws + off;
    off += (bytes + 255) & ~(size_t)255;
    return p;
  };
  __bf16* hbf   = (__bf16*)alloc((size_t)ROWS * DM * 2);       // 8 MB
  __bf16* WqT   = (__bf16*)alloc((size_t)DM * DM * 2);         // 2 MB
  __bf16* WkvT  = (__bf16*)alloc((size_t)2 * DM * DM * 2);     // 4 MB
  __bf16* WoT   = (__bf16*)alloc((size_t)DM * DM * 2);         // 2 MB
  __bf16* qbf   = (__bf16*)alloc((size_t)ROWS * DM * 2);       // 8 MB
  __bf16* kvbf  = (__bf16*)alloc((size_t)ROWS * 2 * DM * 2);   // 16 MB
  __bf16* Vt    = (__bf16*)alloc((size_t)BATCH * NH * DH * SEQ * 2); // 8 MB
  __bf16* vecbf = (__bf16*)alloc((size_t)ROWS * DM * 2);       // 8 MB

  // 1) fp32 -> bf16 (+ weight transposes)
  cvt_f32_bf16<<<(ROWS * DM + 255) / 256, 256, 0, stream>>>(h, hbf, ROWS * DM);
  cvt_transpose<<<(DM * DM + 255) / 256, 256, 0, stream>>>(Wq, WqT, DM, DM);
  cvt_transpose<<<(2 * DM * DM + 255) / 256, 256, 0, stream>>>(Wkv, WkvT, DM, 2 * DM);
  cvt_transpose<<<(DM * DM + 255) / 256, 256, 0, stream>>>(Wo, WoT, DM, DM);

  // 2) projections (WMMA bf16)
  gemm_bf16<true><<<dim3(DM / 64, ROWS / 64), 128, 0, stream>>>(hbf, WqT, qbf, ROWS, DM, DM);
  gemm_bf16<true><<<dim3(2 * DM / 64, ROWS / 64), 128, 0, stream>>>(hbf, WkvT, kvbf, ROWS, 2 * DM, DM);

  // 3) V -> V^T per (b, head)
  repack_vt<<<(ROWS * DM + 255) / 256, 256, 0, stream>>>(kvbf, Vt);

  // 4) attention (QK^T WMMA -> LDS softmax -> PV WMMA)
  const size_t smem = (size_t)16 * SEQ * 4 + (size_t)16 * SEQ * 2; // 192 KB
  attn_kernel<<<dim3(SEQ / 16, BATCH * NH), 256, smem, stream>>>(qbf, kvbf, Vt, vecbf);

  // 5) output projection, f32 out
  gemm_bf16<false><<<dim3(DM / 64, ROWS / 64), 128, 0, stream>>>(vecbf, WoT, d_out, ROWS, DM, DM);
}
